// TransformerDecoderLayer_5695126634635
// MI455X (gfx1250) — matrix operand order repaired and checked by reading, not compiled, not run on hardware
//
#include <hip/hip_runtime.h>
#include <hip/hip_bf16.h>

typedef __attribute__((ext_vector_type(16))) _Float16 v16h;
typedef __attribute__((ext_vector_type(8)))  _Float16 v8h;
typedef __attribute__((ext_vector_type(8)))  float    v8f;
typedef int v4i_t __attribute__((vector_size(16)));

#define MODEL_DIM 1024
#define HKDIM     1024
#define FF_DIM    4096
#define SEQ       2048
#define BATCH     2
#define NTOK      4096   // BATCH*SEQ
#define HEADS     16
#define KSZ       64

// ---------------------------------------------------------------- helpers
__device__ __forceinline__ v16h cat16(v8h lo, v8h hi) {
  return __builtin_shufflevector(lo, hi, 0,1,2,3,4,5,6,7,8,9,10,11,12,13,14,15);
}

__device__ __forceinline__ v8f wmma_f16(v16h a, v16h b, v8f c) {
  // D = A(16x32 f16) * B(32x16 f16) + C(16x16 f32)
  return __builtin_amdgcn_wmma_f32_16x16x32_f16(false, a, false, b,
                                                (short)0, c, false, false);
}

// ---- CDNA5 async global->LDS copy (ASYNCcnt) ----
// Builtin exists on this toolchain; parameter 0 is (as1) v4i*, LDS side as3.
#if defined(__has_builtin) && __has_builtin(__builtin_amdgcn_global_load_async_to_lds_b128)
#define ASYNC_CP_B128(gp, lp)                                            \
  __builtin_amdgcn_global_load_async_to_lds_b128(                        \
      (__attribute__((address_space(1))) v4i_t*)(gp),                    \
      (__attribute__((address_space(3))) v4i_t*)(lp), 0, 0)
#else
// vdst = LDS byte offset (low 32 bits of generic LDS address per aperture rule)
#define ASYNC_CP_B128(gp, lp)                                            \
  asm volatile("global_load_async_to_lds_b128 %0, %1, off"               \
               :: "v"((unsigned)(size_t)(lp)),                           \
                  "v"((unsigned long long)(size_t)(gp)) : "memory")
#endif

__device__ __forceinline__ void wait_async_lds() {
#if defined(__has_builtin) && __has_builtin(__builtin_amdgcn_s_wait_asynccnt)
  __builtin_amdgcn_s_wait_asynccnt(0);
#else
  asm volatile("s_wait_asynccnt 0x0" ::: "memory");
#endif
}

// ---------------------------------------------------------------- convert
__global__ void f32_to_f16_kernel(const float* __restrict__ src,
                                  _Float16* __restrict__ dst, int n) {
  int i = blockIdx.x * blockDim.x + threadIdx.x;
  int stride = gridDim.x * blockDim.x;
  for (; i < n; i += stride) dst[i] = (_Float16)src[i];
}

// ---------------------------------------------------------------- NT GEMM
// C[M,N] = A[M,K](f16,row-major) * B[N,K](f16,row-major)^T + bias
// Template epilogue: RELU, VT (store f16 transposed V^T), RES (+resid f32),
// OF (f32 out), OH (f16 out). Double-buffered LDS with async global->LDS.
template <bool RELU, bool VT, bool RES, bool OF, bool OH>
__launch_bounds__(256)
__global__ void gemm_nt_wmma(const _Float16* __restrict__ A,
                             const _Float16* __restrict__ B,
                             const float* __restrict__ bias,
                             const float* __restrict__ resid,
                             float* __restrict__ outF,
                             _Float16* __restrict__ outH,
                             int M, int N, int K) {
  constexpr int LDT = 48;                        // padded LDS row stride
  __shared__ alignas(16) _Float16 As[2][128 * LDT];
  __shared__ alignas(16) _Float16 Bs[2][128 * LDT];

  const int t    = threadIdx.x;
  const int w    = t >> 5;        // wave 0..7
  const int lane = t & 31;
  const int half = lane >> 4;
  const int lr   = lane & 15;
  const int mBase = blockIdx.y * 128;
  const int nBase = blockIdx.x * 128;
  const int wm = (w & 3) * 32;    // 4 waves along M
  const int wn = (w >> 2) * 64;   // 2 waves along N

  // per-thread staging coordinates (2 passes of 16B per tile)
  const int srow0 = t >> 2;              // rows 0..63
  const int scol  = (t & 3) * 8;         // cols {0,8,16,24}
  const _Float16* gA0 = A + (size_t)(mBase + srow0) * K + scol;
  const _Float16* gA1 = A + (size_t)(mBase + srow0 + 64) * K + scol;
  const _Float16* gB0 = B + (size_t)(nBase + srow0) * K + scol;
  const _Float16* gB1 = B + (size_t)(nBase + srow0 + 64) * K + scol;

  auto stage = [&](int k0, int buf) {
    _Float16* la0 = &As[buf][srow0 * LDT + scol];
    _Float16* la1 = &As[buf][(srow0 + 64) * LDT + scol];
    _Float16* lb0 = &Bs[buf][srow0 * LDT + scol];
    _Float16* lb1 = &Bs[buf][(srow0 + 64) * LDT + scol];
    ASYNC_CP_B128(gA0 + k0, la0);
    ASYNC_CP_B128(gA1 + k0, la1);
    ASYNC_CP_B128(gB0 + k0, lb0);
    ASYNC_CP_B128(gB1 + k0, lb1);
  };

  v8f acc[2][4];
  #pragma unroll
  for (int i = 0; i < 2; ++i)
    #pragma unroll
    for (int j = 0; j < 4; ++j)
      acc[i][j] = (v8f){0.f,0.f,0.f,0.f,0.f,0.f,0.f,0.f};

  const int nk = K >> 5;
  stage(0, 0);
  for (int kb = 0; kb < nk; ++kb) {
    wait_async_lds();
    __syncthreads();                       // buffer kb&1 ready for all waves
    if (kb + 1 < nk) stage((kb + 1) << 5, (kb + 1) & 1);

    const _Float16* as = &As[kb & 1][0];
    const _Float16* bs = &Bs[kb & 1][0];
    v16h af[2], bf[4];
    #pragma unroll
    for (int i = 0; i < 2; ++i) {
      int row = wm + i * 16 + lr;          // A frag: M=lane, K split by half
      v8h lo = *(const v8h*)(as + row * LDT + 8 * half);
      v8h hi = *(const v8h*)(as + row * LDT + 16 + 8 * half);
      af[i] = cat16(lo, hi);
    }
    #pragma unroll
    for (int j = 0; j < 4; ++j) {
      int row = wn + j * 16 + lr;          // B frag: N=lane, K split by half
      v8h lo = *(const v8h*)(bs + row * LDT + 8 * half);
      v8h hi = *(const v8h*)(bs + row * LDT + 16 + 8 * half);
      bf[j] = cat16(lo, hi);
    }
    #pragma unroll
    for (int i = 0; i < 2; ++i)
      #pragma unroll
      for (int j = 0; j < 4; ++j)
        acc[i][j] = wmma_f16(af[i], bf[j], acc[i][j]);
  }

  // straight-line templated epilogue
  #pragma unroll
  for (int i = 0; i < 2; ++i) {
    #pragma unroll
    for (int j = 0; j < 4; ++j) {
      const int gn = nBase + wn + j * 16 + lr;
      const float bb = bias[gn];
      #pragma unroll
      for (int r = 0; r < 8; ++r) {
        const int gm = mBase + wm + i * 16 + r + 8 * half;  // C: M = r + 8*half
        float v = acc[i][j][r] + bb;
        if constexpr (RELU) v = fmaxf(v, 0.0f);
        if constexpr (RES)  v += resid[(size_t)gm * N + gn];
        if constexpr (OF)   outF[(size_t)gm * N + gn] = v;
        if constexpr (OH) {
          if constexpr (VT) {
            const int b = gm >> 11, s = gm & (SEQ - 1);
            outH[((size_t)((b << 10) + gn)) * SEQ + s] = (_Float16)v;
          } else {
            outH[(size_t)gm * N + gn] = (_Float16)v;
          }
        }
      }
    }
  }
}

// ---------------------------------------------------------------- attention
// Flash-style causal attention. 4 waves/block, each wave owns one 16-row
// q-block of one (b,h); streams 32-wide key blocks with online softmax.
__launch_bounds__(128)
__global__ void attn_wmma(const _Float16* __restrict__ Q,
                          const _Float16* __restrict__ Km,
                          const _Float16* __restrict__ Vt,
                          _Float16* __restrict__ O) {
  __shared__ alignas(16) _Float16 Ptile[4][16 * 32];
  const int t    = threadIdx.x;
  const int w    = t >> 5;
  const int lane = t & 31;
  const int half = lane >> 4;
  const int lr   = lane & 15;
  const int h    = blockIdx.y;
  const int b    = blockIdx.z;
  const int qbase = (blockIdx.x * 4 + w) * 16;
  const int tok0  = b * SEQ;

  // Q A-fragments (16 rows x 64 features -> two 16x32 frags)
  v16h aq[2];
  {
    const _Float16* qp = Q + (size_t)(tok0 + qbase + lr) * HKDIM + h * KSZ;
    #pragma unroll
    for (int f = 0; f < 2; ++f) {
      v8h lo = *(const v8h*)(qp + f * 32 + 8 * half);
      v8h hi = *(const v8h*)(qp + f * 32 + 16 + 8 * half);
      aq[f] = cat16(lo, hi);
    }
  }

  v8f o[4];
  #pragma unroll
  for (int n = 0; n < 4; ++n) o[n] = (v8f){0.f,0.f,0.f,0.f,0.f,0.f,0.f,0.f};
  float mrun[8], lrun[8];
  #pragma unroll
  for (int r = 0; r < 8; ++r) { mrun[r] = -1e30f; lrun[r] = 0.0f; }

  const int jmax = (qbase + 15) >> 5;
  for (int jb = 0; jb <= jmax; ++jb) {
    const int jbase = jb * 32;

    // scores: Q[16x64] * K_blk^T -> two 16x16 tiles (tokens jbase..+31)
    v8f s0 = (v8f){0.f,0.f,0.f,0.f,0.f,0.f,0.f,0.f};
    v8f s1 = s0;
    #pragma unroll
    for (int n = 0; n < 2; ++n) {
      const _Float16* kp = Km + (size_t)(tok0 + jbase + n * 16 + lr) * HKDIM + h * KSZ;
      #pragma unroll
      for (int f = 0; f < 2; ++f) {
        v8h lo = *(const v8h*)(kp + f * 32 + 8 * half);
        v8h hi = *(const v8h*)(kp + f * 32 + 16 + 8 * half);
        v16h bk = cat16(lo, hi);
        if (n == 0) s0 = wmma_f16(aq[f], bk, s0);
        else        s1 = wmma_f16(aq[f], bk, s1);
      }
    }

    // scale + causal mask (element: row m = qbase+r+8*half, col = token)
    const int col0 = jbase + lr, col1 = jbase + 16 + lr;
    #pragma unroll
    for (int r = 0; r < 8; ++r) {
      int m = qbase + r + 8 * half;
      float x0 = s0[r] * 0.125f; if (col0 > m) x0 = -1e30f;
      float x1 = s1[r] * 0.125f; if (col1 > m) x1 = -1e30f;
      s0[r] = x0; s1[r] = x1;
    }

    // online softmax (row reduction across 16-lane groups)
    float corr[8];
    #pragma unroll
    for (int r = 0; r < 8; ++r) {
      float rm = fmaxf(s0[r], s1[r]);
      rm = fmaxf(rm, __shfl_xor(rm, 1, 32));
      rm = fmaxf(rm, __shfl_xor(rm, 2, 32));
      rm = fmaxf(rm, __shfl_xor(rm, 4, 32));
      rm = fmaxf(rm, __shfl_xor(rm, 8, 32));
      float mnew = fmaxf(mrun[r], rm);
      float c  = __expf(mrun[r] - mnew);
      float p0 = __expf(s0[r] - mnew);
      float p1 = __expf(s1[r] - mnew);
      float rs = p0 + p1;
      rs += __shfl_xor(rs, 1, 32);
      rs += __shfl_xor(rs, 2, 32);
      rs += __shfl_xor(rs, 4, 32);
      rs += __shfl_xor(rs, 8, 32);
      lrun[r] = lrun[r] * c + rs;
      mrun[r] = mnew;
      corr[r] = c;
      s0[r] = p0; s1[r] = p1;
    }
    #pragma unroll
    for (int n = 0; n < 4; ++n)
      #pragma unroll
      for (int r = 0; r < 8; ++r) o[n][r] *= corr[r];

    // C-tile layout -> A-fragment layout via per-wave LDS tile
    _Float16* P = &Ptile[w][0];
    #pragma unroll
    for (int r = 0; r < 8; ++r) {
      int m = r + 8 * half;
      P[m * 32 + lr]      = (_Float16)s0[r];
      P[m * 32 + 16 + lr] = (_Float16)s1[r];
    }
    asm volatile("s_wait_dscnt 0" ::: "memory");   // same-wave LDS turnaround
    v16h pf;
    {
      v8h lo = *(const v8h*)(P + lr * 32 + 8 * half);
      v8h hi = *(const v8h*)(P + lr * 32 + 16 + 8 * half);
      pf = cat16(lo, hi);
    }

    // O += P[16x32] * V_blk[32x64]  (V^T layout -> contiguous token runs)
    #pragma unroll
    for (int n = 0; n < 4; ++n) {
      const _Float16* vp = Vt +
          ((size_t)((b << 10) + h * KSZ + n * 16 + lr)) * SEQ + jbase + 8 * half;
      v16h bv = cat16(*(const v8h*)(vp), *(const v8h*)(vp + 16));
      o[n] = wmma_f16(pf, bv, o[n]);
    }
  }

  // normalize + store
  #pragma unroll
  for (int r = 0; r < 8; ++r) {
    float inv = 1.0f / lrun[r];
    int tok = tok0 + qbase + r + 8 * half;
    #pragma unroll
    for (int n = 0; n < 4; ++n)
      O[(size_t)tok * HKDIM + h * KSZ + n * 16 + lr] = (_Float16)(o[n][r] * inv);
  }
}

// ---------------------------------------------------------------- launch
extern "C" void kernel_launch(void* const* d_in, const int* in_sizes, int n_in,
                              void* d_out, int out_size, void* d_ws, size_t ws_size,
                              hipStream_t stream) {
  const float* x  = (const float*)d_in[0];
  const float* Wq = (const float*)d_in[1];
  const float* bq = (const float*)d_in[2];
  const float* Wk = (const float*)d_in[3];
  const float* bk = (const float*)d_in[4];
  const float* Wv = (const float*)d_in[5];
  const float* bv = (const float*)d_in[6];
  const float* Wo = (const float*)d_in[7];
  const float* bo = (const float*)d_in[8];
  const float* W1 = (const float*)d_in[9];
  const float* b1 = (const float*)d_in[10];
  const float* W2 = (const float*)d_in[11];
  const float* b2 = (const float*)d_in[12];
  float* out = (float*)d_out;

  char* ws = (char*)d_ws;
  size_t off = 0;
  auto alloc = [&](size_t bytes) -> char* {
    char* p = ws + off;
    off += (bytes + 255) & ~(size_t)255;
    return p;
  };

  _Float16* x16  = (_Float16*)alloc((size_t)NTOK * MODEL_DIM * 2);
  _Float16* wq16 = (_Float16*)alloc((size_t)HKDIM * MODEL_DIM * 2);
  _Float16* wk16 = (_Float16*)alloc((size_t)HKDIM * MODEL_DIM * 2);
  _Float16* wv16 = (_Float16*)alloc((size_t)HKDIM * MODEL_DIM * 2);
  _Float16* wo16 = (_Float16*)alloc((size_t)MODEL_DIM * HKDIM * 2);
  _Float16* w116 = (_Float16*)alloc((size_t)FF_DIM * MODEL_DIM * 2);
  _Float16* w216 = (_Float16*)alloc((size_t)MODEL_DIM * FF_DIM * 2);
  _Float16* q16  = (_Float16*)alloc((size_t)NTOK * HKDIM * 2);
  _Float16* k16  = (_Float16*)alloc((size_t)NTOK * HKDIM * 2);
  _Float16* vt16 = (_Float16*)alloc((size_t)NTOK * HKDIM * 2);   // [B*H*K][S]
  _Float16* o16  = (_Float16*)alloc((size_t)NTOK * HKDIM * 2);
  float*    x2f  = (float*)   alloc((size_t)NTOK * MODEL_DIM * 4);
  _Float16* x2h  = (_Float16*)alloc((size_t)NTOK * MODEL_DIM * 2);
  _Float16* h16  = (_Float16*)alloc((size_t)NTOK * FF_DIM * 2);

  // f32 -> f16 conversions
  f32_to_f16_kernel<<<1024, 256, 0, stream>>>(x,  x16,  NTOK * MODEL_DIM);
  f32_to_f16_kernel<<<1024, 256, 0, stream>>>(Wq, wq16, HKDIM * MODEL_DIM);
  f32_to_f16_kernel<<<1024, 256, 0, stream>>>(Wk, wk16, HKDIM * MODEL_DIM);
  f32_to_f16_kernel<<<1024, 256, 0, stream>>>(Wv, wv16, HKDIM * MODEL_DIM);
  f32_to_f16_kernel<<<1024, 256, 0, stream>>>(Wo, wo16, MODEL_DIM * HKDIM);
  f32_to_f16_kernel<<<1024, 256, 0, stream>>>(W1, w116, FF_DIM * MODEL_DIM);
  f32_to_f16_kernel<<<1024, 256, 0, stream>>>(W2, w216, MODEL_DIM * FF_DIM);

  dim3 gP(HKDIM / 128, NTOK / 128);        // (8, 32)
  // Q / K projections (f16 out only)
  gemm_nt_wmma<false,false,false,false,true><<<gP, 256, 0, stream>>>(
      x16, wq16, bq, nullptr, nullptr, q16, NTOK, HKDIM, MODEL_DIM);
  gemm_nt_wmma<false,false,false,false,true><<<gP, 256, 0, stream>>>(
      x16, wk16, bk, nullptr, nullptr, k16, NTOK, HKDIM, MODEL_DIM);
  // V projection, stored transposed for the PV gemm
  gemm_nt_wmma<false,true,false,false,true><<<gP, 256, 0, stream>>>(
      x16, wv16, bv, nullptr, nullptr, vt16, NTOK, HKDIM, MODEL_DIM);

  // flash attention
  attn_wmma<<<dim3(SEQ / 64, HEADS, BATCH), 128, 0, stream>>>(q16, k16, vt16, o16);

  // output projection + residual: x2 = x + O*Wo^T + bo  (f32 + f16 copies)
  gemm_nt_wmma<false,false,true,true,true><<<gP, 256, 0, stream>>>(
      o16, wo16, bo, x, x2f, x2h, NTOK, MODEL_DIM, HKDIM);

  // FFN up: h = relu(x2*W1^T + b1)
  dim3 gF(FF_DIM / 128, NTOK / 128);       // (32, 32)
  gemm_nt_wmma<true,false,false,false,true><<<gF, 256, 0, stream>>>(
      x2h, w116, b1, nullptr, nullptr, h16, NTOK, FF_DIM, MODEL_DIM);

  // FFN down + residual: out = x2 + h*W2^T + b2
  gemm_nt_wmma<false,false,true,true,false><<<gP, 256, 0, stream>>>(
      h16, w216, b2, x2f, out, nullptr, NTOK, MODEL_DIM, FF_DIM);
}
